// ConvAttBlock_3143916061264
// MI455X (gfx1250) — matrix operand order repaired and checked by reading, not compiled, not run on hardware
//
#include <hip/hip_runtime.h>

#define CIN 192
#define C384 384
#define NHEADS 12
#define HD 32
#define NWIN 16
#define NTOK 50
#define MLOC 51200   // 1024 windows * 50 tokens
#define MREG 1024    // 64 images * 16 windows

typedef __attribute__((ext_vector_type(16))) _Float16 v16h;
typedef __attribute__((ext_vector_type(8)))  _Float16 v8h;
typedef __attribute__((ext_vector_type(8)))  float    v8f;

__device__ __forceinline__ float gelu_exact(float x) {
    return 0.5f * x * (1.0f + erff(x * 0.70710678118654752f));
}

// ---------------------------------------------------------------------------
// Pack f32 weight [K,N] into f16 WMMA B-fragment order:
//   o[((tn*Kb + kb)*32 + lane)*16 + h] = W[kb*32 + (h&8)*2 + (lane>>4)*8 + (h&7)][tn*16 + (lane&15)]
// so each lane's 16-half fragment is 32 contiguous bytes (two b128 loads).
// ---------------------------------------------------------------------------
__global__ void pack_w16(const float* __restrict__ w, _Float16* __restrict__ o, int K, int N) {
    int idx = blockIdx.x * blockDim.x + threadIdx.x;
    if (idx >= K * N) return;
    int h = idx & 15; int r = idx >> 4;
    int lane = r & 31; r >>= 5;
    int Kb = K >> 5;
    int kb = r % Kb; int tn = r / Kb;
    int k = kb * 32 + (h & 8) * 2 + (lane >> 4) * 8 + (h & 7);
    int n = tn * 16 + (lane & 15);
    o[idx] = (_Float16)w[(size_t)k * N + n];
}

// ---------------------------------------------------------------------------
// Per-pixel LayerNorm stats over 192 channels of an NCHW tensor.
// ---------------------------------------------------------------------------
__global__ void __launch_bounds__(64) ln_stats_nchw(const float* __restrict__ x, int HW,
                                                    float* __restrict__ stats) {
    int p = blockIdx.x;
    int b = p / HW, pix = p % HW;
    const float* xp = x + (size_t)b * CIN * HW + pix;
    __shared__ float red[64];
    int t = threadIdx.x;
    float v[3];
    float s = 0.f;
#pragma unroll
    for (int j = 0; j < 3; ++j) { v[j] = xp[(size_t)(t + j * 64) * HW]; s += v[j]; }
    red[t] = s; __syncthreads();
    for (int off = 32; off > 0; off >>= 1) { if (t < off) red[t] += red[t + off]; __syncthreads(); }
    float mean = red[0] / (float)CIN; __syncthreads();
    float ss = 0.f;
#pragma unroll
    for (int j = 0; j < 3; ++j) { float d = v[j] - mean; ss += d * d; }
    red[t] = ss; __syncthreads();
    for (int off = 32; off > 0; off >>= 1) { if (t < off) red[t] += red[t + off]; __syncthreads(); }
    if (t == 0) {
        stats[2 * (size_t)p]     = mean;
        stats[2 * (size_t)p + 1] = rsqrtf(red[0] / (float)CIN + 1e-5f);
    }
}

// ---------------------------------------------------------------------------
// Grouped stride-2 conv (3x3, pad 1, 192 groups) fused with LN + exact GELU,
// writing straight into the windowed token layout stream[win][tok][384].
// ---------------------------------------------------------------------------
__global__ void proj_conv_win(const float* __restrict__ x, const float* __restrict__ stats,
                              const float* __restrict__ g, const float* __restrict__ bta,
                              const float* __restrict__ w, const float* __restrict__ cb,
                              float* __restrict__ stream, int Hin, int is_cls, int total) {
    int gid = blockIdx.x * blockDim.x + threadIdx.x;
    if (gid >= total) return;
    int oc = gid % C384; int r = gid / C384;
    int Hout = Hin >> 1;
    int ow = r % Hout; r /= Hout;
    int oh = r % Hout; int b = r / Hout;
    int ic = oc >> 1;
    float acc = 0.f;
#pragma unroll
    for (int kh = 0; kh < 3; ++kh) {
        int ih = 2 * oh - 1 + kh;
        if (ih < 0 || ih >= Hin) continue;
#pragma unroll
        for (int kw = 0; kw < 3; ++kw) {
            int iw = 2 * ow - 1 + kw;
            if (iw < 0 || iw >= Hin) continue;
            float xv = x[(((size_t)b * CIN + ic) * Hin + ih) * Hin + iw];
            const float* st = stats + 2 * ((size_t)b * Hin * Hin + (size_t)ih * Hin + iw);
            float v = (xv - st[0]) * st[1] * g[ic] + bta[ic];
            v = gelu_exact(v);
            acc += v * w[(kh * 3 + kw) * C384 + oc];
        }
    }
    acc += cb[oc];
    int win, tok;
    if (is_cls) { win = b * NWIN + oh * 4 + ow; tok = 0; }
    else        { win = b * NWIN + (oh / 7) * 4 + (ow / 7); tok = 1 + (oh % 7) * 7 + (ow % 7); }
    stream[((size_t)win * NTOK + tok) * C384 + oc] = acc;
}

// ---------------------------------------------------------------------------
// LayerNorm over 384 channels -> f16 output.
// ---------------------------------------------------------------------------
__global__ void __launch_bounds__(128) ln_to_f16(const float* __restrict__ x, long long rstride,
                                                 const float* __restrict__ g, const float* __restrict__ b,
                                                 _Float16* __restrict__ y) {
    int row = blockIdx.x;
    const float* xr = x + (size_t)row * rstride;
    __shared__ float red[128];
    int t = threadIdx.x;
    float v[3];
    float s = 0.f;
#pragma unroll
    for (int j = 0; j < 3; ++j) { v[j] = xr[t + j * 128]; s += v[j]; }
    red[t] = s; __syncthreads();
    for (int off = 64; off > 0; off >>= 1) { if (t < off) red[t] += red[t + off]; __syncthreads(); }
    float mean = red[0] / (float)C384; __syncthreads();
    float ss = 0.f;
#pragma unroll
    for (int j = 0; j < 3; ++j) { float d = v[j] - mean; ss += d * d; }
    red[t] = ss; __syncthreads();
    for (int off = 64; off > 0; off >>= 1) { if (t < off) red[t] += red[t + off]; __syncthreads(); }
    float rstd = rsqrtf(red[0] / (float)C384 + 1e-5f);
    __syncthreads();
#pragma unroll
    for (int j = 0; j < 3; ++j) {
        int c = t + j * 128;
        y[(size_t)row * C384 + c] = (_Float16)((v[j] - mean) * rstd * g[c] + b[c]);
    }
}

// ---------------------------------------------------------------------------
// WMMA GEMM: C[M, N] = A[M,K](f16) @ W(packed f16) + bias [+GELU] [+resid f32]
// One wave computes a 32x64 strip: two A fragments x four B fragments =
// 8 WMMAs per K-step (12 b128 loads), maximizing operand reuse.
// A: lane L = row M=L%16; halves 0..7 -> K=(L/16)*8+h, 8..15 -> K=16+(L/16)*8+(h-8)
// B: packed so each lane's fragment is 32 contiguous bytes.
// C: lane L, vgpr r -> M=(L/16)*8+r, N=L%16.
// ---------------------------------------------------------------------------
__global__ void __launch_bounds__(128)
gemm_wmma_f16(const _Float16* __restrict__ A, const _Float16* __restrict__ Wp,
              const float* __restrict__ bias,
              float* __restrict__ resid, long long res_stride,
              _Float16* __restrict__ out16,
              int M, int N, int K, int gelu_flag) {
    int wave = threadIdx.x >> 5;
    int lane = threadIdx.x & 31;
    int Ng = N >> 6;                       // groups of 4 N-tiles
    int ntot = (M >> 5) * Ng;              // 32-row strips
    int tile = blockIdx.x * 4 + wave;
    if (tile >= ntot) return;              // wave-uniform: EXEC stays all-ones
    int tm = tile / Ng, tg = tile - tm * Ng;
    int lmod = lane & 15, lhi = lane >> 4;
    int Kb = K >> 5;

    v8f acc00 = {}, acc01 = {}, acc02 = {}, acc03 = {};
    v8f acc10 = {}, acc11 = {}, acc12 = {}, acc13 = {};
    const _Float16* arow0 = A + (size_t)(tm * 32 + lmod) * K + lhi * 8;
    const _Float16* arow1 = arow0 + (size_t)16 * K;
    const _Float16* bbase = Wp + ((size_t)(tg * 4) * Kb * 32 + lane) * 16;
    const size_t ts = (size_t)Kb * 512;    // halves per packed N-tile

    for (int kb = 0; kb < Kb; ++kb) {
        v16h a0, a1, b0, b1, b2, b3;
        ((v8h*)&a0)[0] = *(const v8h*)(arow0 + kb * 32);
        ((v8h*)&a0)[1] = *(const v8h*)(arow0 + kb * 32 + 16);
        ((v8h*)&a1)[0] = *(const v8h*)(arow1 + kb * 32);
        ((v8h*)&a1)[1] = *(const v8h*)(arow1 + kb * 32 + 16);
        const _Float16* bp = bbase + (size_t)kb * 512;
        ((v8h*)&b0)[0] = *(const v8h*)(bp);
        ((v8h*)&b0)[1] = *(const v8h*)(bp + 8);
        ((v8h*)&b1)[0] = *(const v8h*)(bp + ts);
        ((v8h*)&b1)[1] = *(const v8h*)(bp + ts + 8);
        ((v8h*)&b2)[0] = *(const v8h*)(bp + 2 * ts);
        ((v8h*)&b2)[1] = *(const v8h*)(bp + 2 * ts + 8);
        ((v8h*)&b3)[0] = *(const v8h*)(bp + 3 * ts);
        ((v8h*)&b3)[1] = *(const v8h*)(bp + 3 * ts + 8);
        acc00 = __builtin_amdgcn_wmma_f32_16x16x32_f16(false, a0, false, b0, (short)0, acc00, false, false);
        acc01 = __builtin_amdgcn_wmma_f32_16x16x32_f16(false, a0, false, b1, (short)0, acc01, false, false);
        acc02 = __builtin_amdgcn_wmma_f32_16x16x32_f16(false, a0, false, b2, (short)0, acc02, false, false);
        acc03 = __builtin_amdgcn_wmma_f32_16x16x32_f16(false, a0, false, b3, (short)0, acc03, false, false);
        acc10 = __builtin_amdgcn_wmma_f32_16x16x32_f16(false, a1, false, b0, (short)0, acc10, false, false);
        acc11 = __builtin_amdgcn_wmma_f32_16x16x32_f16(false, a1, false, b1, (short)0, acc11, false, false);
        acc12 = __builtin_amdgcn_wmma_f32_16x16x32_f16(false, a1, false, b2, (short)0, acc12, false, false);
        acc13 = __builtin_amdgcn_wmma_f32_16x16x32_f16(false, a1, false, b3, (short)0, acc13, false, false);
    }

    v8f accs[2][4] = {{acc00, acc01, acc02, acc03}, {acc10, acc11, acc12, acc13}};
#pragma unroll
    for (int i = 0; i < 2; ++i) {
#pragma unroll
        for (int j = 0; j < 4; ++j) {
#pragma unroll
            for (int r = 0; r < 8; ++r) {
                int mm = tm * 32 + i * 16 + lhi * 8 + r;
                int nn = (tg * 4 + j) * 16 + lmod;
                float v = accs[i][j][r];
                if (bias) v += bias[nn];
                if (gelu_flag) v = gelu_exact(v);
                if (resid) {
                    float* rp = resid + (size_t)mm * res_stride + nn;
                    v += *rp;
                    *rp = v;
                }
                if (out16) out16[(size_t)mm * N + nn] = (_Float16)v;
            }
        }
    }
}

// ---------------------------------------------------------------------------
// Regional attention: per (image, head), 16 tokens, HD=32, no rel-pos bias.
// ---------------------------------------------------------------------------
__global__ void __launch_bounds__(256) attn_reg(const _Float16* __restrict__ qkv,
                                                _Float16* __restrict__ att) {
    int b = blockIdx.x, h = blockIdx.y;
    __shared__ _Float16 sq[16 * HD], sk[16 * HD], sv[16 * HD];
    __shared__ float sc[16 * 16];
    int t = threadIdx.x;
    for (int i = t; i < 16 * HD; i += 256) {
        int tok = i >> 5, d = i & 31;
        size_t base = (size_t)(b * NWIN + tok) * 1152 + h * HD + d;
        sq[i] = qkv[base];
        sk[i] = qkv[base + C384];
        sv[i] = qkv[base + 2 * C384];
    }
    __syncthreads();
    const float scale = 0.17677669529663687f;  // 32^-0.5
    {
        int q = t >> 4, k = t & 15;
        float a = 0.f;
#pragma unroll
        for (int d = 0; d < HD; ++d) a += (float)sq[q * HD + d] * (float)sk[k * HD + d];
        sc[t] = a * scale;
    }
    __syncthreads();
    if (t < 16) {
        float mx = -1e30f;
        for (int k = 0; k < 16; ++k) mx = fmaxf(mx, sc[t * 16 + k]);
        float sum = 0.f;
        for (int k = 0; k < 16; ++k) { float e = expf(sc[t * 16 + k] - mx); sc[t * 16 + k] = e; sum += e; }
        float inv = 1.f / sum;
        for (int k = 0; k < 16; ++k) sc[t * 16 + k] *= inv;
    }
    __syncthreads();
    for (int i = t; i < 16 * HD; i += 256) {
        int q = i >> 5, d = i & 31;
        float a = 0.f;
#pragma unroll
        for (int k = 0; k < 16; ++k) a += sc[q * 16 + k] * (float)sv[k * HD + d];
        att[(size_t)(b * NWIN + q) * C384 + h * HD + d] = (_Float16)a;
    }
}

// ---------------------------------------------------------------------------
// Local attention: per (window, head), 50 tokens, rel-pos bias on patch-patch.
// ---------------------------------------------------------------------------
__global__ void __launch_bounds__(256) attn_local(const _Float16* __restrict__ qkv,
                                                  const float* __restrict__ relpos,
                                                  _Float16* __restrict__ att) {
    int win = blockIdx.x, h = blockIdx.y;
    __shared__ _Float16 sq[NTOK * HD], sk[NTOK * HD], sv[NTOK * HD];
    __shared__ float sc[NTOK * NTOK];
    int t = threadIdx.x;
    for (int i = t; i < NTOK * HD; i += 256) {
        int tok = i >> 5, d = i & 31;
        size_t base = ((size_t)win * NTOK + tok) * 1152 + h * HD + d;
        sq[i] = qkv[base];
        sk[i] = qkv[base + C384];
        sv[i] = qkv[base + 2 * C384];
    }
    __syncthreads();
    const float scale = 0.17677669529663687f;
    for (int i = t; i < NTOK * NTOK; i += 256) {
        int q = i / NTOK, k = i % NTOK;
        float a = 0.f;
#pragma unroll
        for (int d = 0; d < HD; ++d) a += (float)sq[q * HD + d] * (float)sk[k * HD + d];
        a *= scale;
        if (q > 0 && k > 0) {
            int qi = q - 1, ki = k - 1;
            int qr = qi / 7, qc = qi % 7, kr = ki / 7, kc = ki % 7;
            a += relpos[h * 169 + (qr - kr + 6) * 13 + (qc - kc + 6)];
        }
        sc[i] = a;
    }
    __syncthreads();
    if (t < NTOK) {
        float mx = -1e30f;
        for (int k = 0; k < NTOK; ++k) mx = fmaxf(mx, sc[t * NTOK + k]);
        float sum = 0.f;
        for (int k = 0; k < NTOK; ++k) { float e = expf(sc[t * NTOK + k] - mx); sc[t * NTOK + k] = e; sum += e; }
        float inv = 1.f / sum;
        for (int k = 0; k < NTOK; ++k) sc[t * NTOK + k] *= inv;
    }
    __syncthreads();
    for (int i = t; i < NTOK * HD; i += 256) {
        int q = i >> 5, d = i & 31;
        float a = 0.f;
        for (int k = 0; k < NTOK; ++k) a += sc[q * NTOK + k] * (float)sv[k * HD + d];
        att[((size_t)win * NTOK + q) * C384 + h * HD + d] = (_Float16)a;
    }
}

// ---------------------------------------------------------------------------
// Fold windowed stream back into cls_out [64,384,4,4] ++ patch_out [64,384,28,28]
// ---------------------------------------------------------------------------
__global__ void scatter_out(const float* __restrict__ stream, float* __restrict__ out, int total) {
    int gid = blockIdx.x * blockDim.x + threadIdx.x;
    if (gid >= total) return;
    if (gid < 64 * C384 * 16) {
        int w = gid % 4; int r = gid / 4;
        int hh = r % 4; r /= 4;
        int c = r % C384; int b = r / C384;
        out[gid] = stream[(size_t)(b * NWIN + hh * 4 + w) * NTOK * C384 + c];
    } else {
        int g2 = gid - 64 * C384 * 16;
        int ow = g2 % 28; int r = g2 / 28;
        int oh = r % 28; r /= 28;
        int c = r % C384; int b = r / C384;
        int win = b * NWIN + (oh / 7) * 4 + (ow / 7);
        int tok = 1 + (oh % 7) * 7 + (ow % 7);
        out[gid] = stream[((size_t)win * NTOK + tok) * C384 + c];
    }
}

// ---------------------------------------------------------------------------
extern "C" void kernel_launch(void* const* d_in, const int* in_sizes, int n_in,
                              void* d_out, int out_size, void* d_ws, size_t ws_size,
                              hipStream_t stream) {
    (void)in_sizes; (void)n_in; (void)ws_size;
    const float* cls_tokens  = (const float*)d_in[0];
    const float* patch_tok   = (const float*)d_in[1];
    const float* proj_ln_g   = (const float*)d_in[2];
    const float* proj_ln_b   = (const float*)d_in[3];
    const float* proj_conv_w = (const float*)d_in[4];
    const float* proj_conv_b = (const float*)d_in[5];
    const float* n0_g = (const float*)d_in[6];
    const float* n0_b = (const float*)d_in[7];
    const float* n1_g = (const float*)d_in[8];
    const float* n1_b = (const float*)d_in[9];
    const float* n2_g = (const float*)d_in[10];
    const float* n2_b = (const float*)d_in[11];
    const float* qkv_w = (const float*)d_in[12];
    const float* qkv_b = (const float*)d_in[13];
    const float* ap_w  = (const float*)d_in[14];
    const float* ap_b  = (const float*)d_in[15];
    const float* rel_pos = (const float*)d_in[16];
    const float* fc1_w = (const float*)d_in[17];
    const float* fc1_b = (const float*)d_in[18];
    const float* fc2_w = (const float*)d_in[19];
    const float* fc2_b = (const float*)d_in[20];

    // ------- workspace layout (byte offsets, all 256-aligned) -------
    char* ws = (char*)d_ws;
    float*     strm    = (float*)ws;                                 // 51200*384 f32   = 78,643,200 B
    _Float16*  buf16a  = (_Float16*)(ws + 78643200);                 // 51200*384 f16   = 39,321,600 B
    _Float16*  qkvhid  = (_Float16*)(ws + 78643200 + 39321600);      // 51200*1536 f16  = 157,286,400 B
    float*     stats   = (float*)(ws + 275251200);                   // 409,600 f32     = 1,638,400 B
    _Float16*  w16     = (_Float16*)(ws + 276889600);                // 3,538,944 f16   = 7,077,888 B
    _Float16*  cls_ln  = (_Float16*)(ws + 283967488);                // 1024*384 f16
    _Float16*  cls_qkv = (_Float16*)(ws + 283967488 + 786432);       // 1024*1152 f16
    _Float16*  cls_att = (_Float16*)(ws + 283967488 + 786432 + 2359296);

    _Float16* qkv16 = w16;                 // 2 * 442368 (packed)
    _Float16* ap16  = w16 + 884736;        // 2 * 147456 (packed)
    _Float16* fc116 = w16 + 1179648;       // 2 * 589824 (packed)
    _Float16* fc216 = w16 + 2359296;       // 2 * 589824 (packed)

    // ------- convert + repack weights into WMMA B-fragment order -------
    for (int i = 0; i < 2; ++i) {
        pack_w16<<<(442368 + 255) / 256, 256, 0, stream>>>(
            qkv_w + (size_t)i * 442368, qkv16 + (size_t)i * 442368, 384, 1152);
        pack_w16<<<(147456 + 255) / 256, 256, 0, stream>>>(
            ap_w + (size_t)i * 147456, ap16 + (size_t)i * 147456, 384, 384);
        pack_w16<<<(589824 + 255) / 256, 256, 0, stream>>>(
            fc1_w + (size_t)i * 589824, fc116 + (size_t)i * 589824, 384, 1536);
        pack_w16<<<(589824 + 255) / 256, 256, 0, stream>>>(
            fc2_w + (size_t)i * 589824, fc216 + (size_t)i * 589824, 1536, 384);
    }

    // ------- projection: LN stats + fused LN/GELU/conv into window layout -------
    float* stats_cls   = stats;              // 64*64*2
    float* stats_patch = stats + 64 * 64 * 2;
    ln_stats_nchw<<<64 * 64, 64, 0, stream>>>(cls_tokens, 64, stats_cls);
    ln_stats_nchw<<<64 * 3136, 64, 0, stream>>>(patch_tok, 3136, stats_patch);
    {
        int tot_c = 64 * 4 * 4 * C384;
        proj_conv_win<<<(tot_c + 255) / 256, 256, 0, stream>>>(
            cls_tokens, stats_cls, proj_ln_g, proj_ln_b, proj_conv_w, proj_conv_b,
            strm, 8, 1, tot_c);
        int tot_p = 64 * 28 * 28 * C384;
        proj_conv_win<<<(tot_p + 255) / 256, 256, 0, stream>>>(
            patch_tok, stats_patch, proj_ln_g, proj_ln_b, proj_conv_w, proj_conv_b,
            strm, 56, 0, tot_p);
    }

    auto ggrid = [](int M, int N) { return ((M >> 5) * (N >> 6) + 3) / 4; };

    for (int i = 0; i < 2; ++i) {
        const _Float16* qw = qkv16 + (size_t)i * 442368;
        const _Float16* aw = ap16  + (size_t)i * 147456;
        const _Float16* w1 = fc116 + (size_t)i * 589824;
        const _Float16* w2 = fc216 + (size_t)i * 589824;

        // ---- regional (cls) attention ----
        ln_to_f16<<<MREG, 128, 0, stream>>>(strm, (long long)(NTOK * C384),
                                            n0_g + i * C384, n0_b + i * C384, cls_ln);
        gemm_wmma_f16<<<ggrid(MREG, 1152), 128, 0, stream>>>(
            cls_ln, qw, qkv_b + i * 1152, nullptr, 0, cls_qkv, MREG, 1152, C384, 0);
        attn_reg<<<dim3(64, NHEADS), 256, 0, stream>>>(cls_qkv, cls_att);
        gemm_wmma_f16<<<ggrid(MREG, C384), 128, 0, stream>>>(
            cls_att, aw, ap_b + i * C384, strm, (long long)(NTOK * C384), nullptr,
            MREG, C384, C384, 0);

        // ---- local (window) attention ----
        ln_to_f16<<<MLOC, 128, 0, stream>>>(strm, (long long)C384,
                                            n1_g + i * C384, n1_b + i * C384, buf16a);
        gemm_wmma_f16<<<ggrid(MLOC, 1152), 128, 0, stream>>>(
            buf16a, qw, qkv_b + i * 1152, nullptr, 0, qkvhid, MLOC, 1152, C384, 0);
        attn_local<<<dim3(1024, NHEADS), 256, 0, stream>>>(
            qkvhid, rel_pos + (size_t)i * NHEADS * 169, buf16a);
        gemm_wmma_f16<<<ggrid(MLOC, C384), 128, 0, stream>>>(
            buf16a, aw, ap_b + i * C384, strm, (long long)C384, nullptr,
            MLOC, C384, C384, 0);

        // ---- MLP ----
        ln_to_f16<<<MLOC, 128, 0, stream>>>(strm, (long long)C384,
                                            n2_g + i * C384, n2_b + i * C384, buf16a);
        gemm_wmma_f16<<<ggrid(MLOC, 1536), 128, 0, stream>>>(
            buf16a, w1, fc1_b + i * 1536, nullptr, 0, qkvhid, MLOC, 1536, C384, 1);
        gemm_wmma_f16<<<ggrid(MLOC, C384), 128, 0, stream>>>(
            qkvhid, w2, fc2_b + i * C384, strm, (long long)C384, nullptr,
            MLOC, C384, 1536, 0);
    }

    // ------- fold back to output layout -------
    scatter_out<<<(out_size + 255) / 256, 256, 0, stream>>>(strm, (float*)d_out, out_size);
}
